// ArcFaceLoss_23922967839356
// MI455X (gfx1250) — matrix operand order repaired and checked by reading, not compiled, not run on hardware
//
#include <hip/hip_runtime.h>
#include <hip/hip_bf16.h>
#include <math.h>

// ---------------- problem constants ----------------
#define BATCH   512
#define EMBED   512
#define NCLS    100000
#define NTILE   64                       // classes per block
#define MTILE   128                      // batch rows per block (8 waves x 16)
#define NBLK    ((NCLS + NTILE - 1) / NTILE)   // 1563
#define CPAD    (NBLK * NTILE)                 // 100032 (zero-padded rows)
#define KCHUNK  64                       // K per LDS stage (2 WMMA K-steps)
#define NCHUNK  (EMBED / KCHUNK)         // 8
#define SCALE_F 64.0f

// ArcFace margin m = 0.5
#define COS_M_F 0.8775825618903728f      // cos(0.5)
#define SIN_M_F 0.479425538604203f       // sin(0.5)
#define TH_F   (-0.8775825618903728f)    // cos(pi - 0.5)
#define MM_F    0.2397127693021015f      // sin(pi - 0.5) * 0.5

typedef _Float16 half8  __attribute__((ext_vector_type(8)));
typedef _Float16 half16 __attribute__((ext_vector_type(16)));
typedef float    float8 __attribute__((ext_vector_type(8)));
typedef int      v4i    __attribute__((vector_size(16)));   // matches builtin param

#define AS1 __attribute__((address_space(1)))
#define AS3 __attribute__((address_space(3)))

#if defined(__has_builtin)
#if __has_builtin(__builtin_amdgcn_global_load_async_to_lds_b128)
#define HAVE_ASYNC_LDS 1
#endif
#endif
#ifndef HAVE_ASYNC_LDS
#define HAVE_ASYNC_LDS 0
#endif

#if defined(__has_builtin) && __has_builtin(__builtin_amdgcn_s_wait_asynccnt)
#define WAIT_ASYNC(n) __builtin_amdgcn_s_wait_asynccnt(n)
#else
#define WAIT_ASYNC(n) asm volatile("s_wait_asynccnt %0" ::"n"(n) : "memory")
#endif

// ---------------------------------------------------------------------------
// Kernel 1: per-row L2 normalize fp32 -> f16 (rows >= `rows` are zero pad).
// One block (256 threads) per row of EMBED=512 elements.
// ---------------------------------------------------------------------------
__global__ __launch_bounds__(256)
void arcface_nrm_f16(const float* __restrict__ src, _Float16* __restrict__ dst,
                     int rows) {
    __shared__ float red[256];
    const int r = blockIdx.x;
    const int t = threadIdx.x;
    _Float16* drow = dst + (size_t)r * EMBED;
    if (r >= rows) {                      // zero the padded class rows
        drow[t]       = (_Float16)0.0f;
        drow[t + 256] = (_Float16)0.0f;
        return;
    }
    const float* srow = src + (size_t)r * EMBED;
    const float a0 = srow[t];
    const float a1 = srow[t + 256];
    red[t] = a0 * a0 + a1 * a1;
    __syncthreads();
    for (int s = 128; s > 0; s >>= 1) {
        if (t < s) red[t] += red[t + s];
        __syncthreads();
    }
    const float rn = 1.0f / fmaxf(sqrtf(red[0]), 1e-12f);
    drow[t]       = (_Float16)(a0 * rn);
    drow[t + 256] = (_Float16)(a1 * rn);
}

// ---------------------------------------------------------------------------
// Stage one thread's 32-byte slice of the B (weight) tile into LDS.
// With async-to-LDS: no VGPR involvement, tracked by ASYNCcnt.
// ---------------------------------------------------------------------------
__device__ __forceinline__
void stage_b_slice(const _Float16* __restrict__ g, _Float16* l) {
#if HAVE_ASYNC_LDS
    __builtin_amdgcn_global_load_async_to_lds_b128(
        (AS1 v4i*)g, (AS3 v4i*)l, 0, 0);
    __builtin_amdgcn_global_load_async_to_lds_b128(
        (AS1 v4i*)(g + 8), (AS3 v4i*)(l + 8), 0, 0);
#else
    *(half8*)(l)     = *(const half8*)(g);
    *(half8*)(l + 8) = *(const half8*)(g + 8);
#endif
}

// ---------------------------------------------------------------------------
// Kernel 2: cosine GEMM via v_wmma_f32_16x16x32_f16 fused with exp(64*cos)
// partial row-sums.  Grid: (NBLK, BATCH/MTILE). Block: 256 threads = 8 waves.
// B (weight) tile [64 classes x 64 K] is double-buffered through LDS via
// async load-to-LDS so all 8 waves share one copy; A fragments come straight
// from the L2-resident f16 x-matrix.  Wave w owns a 16(M) x 64(N) strip.
// ---------------------------------------------------------------------------
__global__ __launch_bounds__(256)
void arcface_gemm_exp(const _Float16* __restrict__ xh,
                      const _Float16* __restrict__ wh,
                      float* __restrict__ partial) {
    __shared__ _Float16 bt[2][NTILE * KCHUNK];   // 2 x 8 KB

    const int t    = threadIdx.x;
    const int lane = t & 31;
    const int wv   = t >> 5;                      // wave 0..7
    const int nb   = blockIdx.x;                  // class tile
    const int mb   = blockIdx.y;                  // batch tile
    const int m0   = mb * MTILE + wv * 16;        // this wave's 16 batch rows
    const int c0   = nb * NTILE;                  // 64 classes
    const int l15  = lane & 15;
    const int h    = lane >> 4;                   // half-wave select

    // B staging: thread covers 32 bytes (16 halfs) of row (t>>2), seg (t&3)
    const int brow = t >> 2;
    const int bseg = t & 3;
    const _Float16* gB = wh + (size_t)(c0 + brow) * EMBED + bseg * 16;
    _Float16* lBa = &bt[0][brow * KCHUNK + bseg * 16];
    _Float16* lBb = &bt[1][brow * KCHUNK + bseg * 16];

    float8 acc[4] = {float8{}, float8{}, float8{}, float8{}};

    // A (16x32 f16): lane holds row M=l15; halfs [k+16h..+15] contiguous
    const _Float16* aBase = xh + (size_t)(m0 + l15) * EMBED + 16 * h;

    stage_b_slice(gB, lBa);                       // chunk 0 in flight

    for (int ck = 0; ck < NCHUNK; ++ck) {
        const int k = ck * KCHUNK;
        // all waves done reading the buffer we are about to overwrite
        __syncthreads();
        if (ck + 1 < NCHUNK) {
            stage_b_slice(gB + k + KCHUNK, (ck & 1) ? lBa : lBb);
#if HAVE_ASYNC_LDS
            WAIT_ASYNC(2);                        // chunk ck landed (in-order)
#endif
        } else {
#if HAVE_ASYNC_LDS
            WAIT_ASYNC(0);
#endif
        }
        __syncthreads();                          // whole tile visible
        const _Float16* lbuf = bt[ck & 1];

        __builtin_prefetch(aBase + k + KCHUNK, 0, 1);  // next A chunk

#pragma unroll
        for (int s = 0; s < 2; ++s) {             // two K=32 WMMA steps
            const half8 alo = *(const half8*)(aBase + k + 32 * s);
            const half8 ahi = *(const half8*)(aBase + k + 32 * s + 8);
            const half16 a = __builtin_shufflevector(
                alo, ahi, 0, 1, 2, 3, 4, 5, 6, 7, 8, 9, 10, 11, 12, 13, 14, 15);
#pragma unroll
            for (int n = 0; n < 4; ++n) {
                // B (32x16): lane = column c0+n*16+l15; 16 contiguous halfs
                const _Float16* bq =
                    lbuf + (n * 16 + l15) * KCHUNK + 32 * s + 16 * h;
                const half8 blo = *(const half8*)(bq);
                const half8 bhi = *(const half8*)(bq + 8);
                const half16 b = __builtin_shufflevector(
                    blo, bhi, 0, 1, 2, 3, 4, 5, 6, 7, 8, 9, 10, 11, 12, 13, 14, 15);
                acc[n] = __builtin_amdgcn_wmma_f32_16x16x32_f16(
                    false, a, false, b, (short)0, acc[n], false, false);
            }
        }
    }

    // Epilogue: C/D layout -> lane holds (M = v + 8h, N = l15) for VGPR v.
#pragma unroll
    for (int v = 0; v < 8; ++v) {
        float e = 0.0f;
#pragma unroll
        for (int n = 0; n < 4; ++n) {
            const int c = c0 + n * 16 + l15;
            const float z = SCALE_F * acc[n][v];
            e += (c < NCLS) ? __expf(z) : 0.0f;   // mask padded classes
        }
        e += __shfl_xor(e, 1);
        e += __shfl_xor(e, 2);
        e += __shfl_xor(e, 4);
        e += __shfl_xor(e, 8);
        if (l15 == 0) {
            const int row = m0 + v + 8 * h;
            partial[(size_t)nb * BATCH + row] = e;
        }
    }
}

// ---------------------------------------------------------------------------
// Kernel 3: cos_t[b] = dot(xh[b], wh[target[b]]) in f32 (consistent with the
// f16 GEMM so the target-column exp swap cancels cleanly).
// ---------------------------------------------------------------------------
__global__ __launch_bounds__(256)
void arcface_cost(const _Float16* __restrict__ xh,
                  const _Float16* __restrict__ wh,
                  const int* __restrict__ target,
                  float* __restrict__ cost) {
    __shared__ float red[256];
    const int b = blockIdx.x;
    const int t = threadIdx.x;
    const int tgt = target[b];
    const _Float16* xr = xh + (size_t)b * EMBED;
    const _Float16* wr = wh + (size_t)tgt * EMBED;
    red[t] = (float)xr[t] * (float)wr[t] +
             (float)xr[t + 256] * (float)wr[t + 256];
    __syncthreads();
    for (int s = 128; s > 0; s >>= 1) {
        if (t < s) red[t] += red[t + s];
        __syncthreads();
    }
    if (t == 0) cost[b] = red[0];
}

// ---------------------------------------------------------------------------
// Kernel 4: per-row logsumexp fix-up + margin math + mean.  One block, 512
// threads; thread b reduces the 1563 class-tile partials (coalesced).
// ---------------------------------------------------------------------------
__global__ __launch_bounds__(512)
void arcface_finalize(const float* __restrict__ partial,
                      const float* __restrict__ cost,
                      float* __restrict__ out) {
    __shared__ float red[512];
    const int b = threadIdx.x;
    float S = 0.0f;
    for (int nb = 0; nb < NBLK; ++nb)
        S += partial[(size_t)nb * BATCH + b];

    const float c   = cost[b];
    const float sn  = sqrtf(fmaxf(1.0f - c * c, 0.0f));
    float phi = c * COS_M_F - sn * SIN_M_F;
    if (!(c > TH_F)) phi = c - MM_F;

    // swap target column's exp(64*cos) for exp(64*phi); 64*|z| <= 64 so no
    // overflow in f32 (e^64 ~ 6.2e27, <=1e5 terms) -> no max-subtraction pass.
    float adj = S - __expf(SCALE_F * c) + __expf(SCALE_F * phi);
    adj = fmaxf(adj, 1e-30f);
    red[b] = logf(adj) - SCALE_F * phi;
    __syncthreads();
    for (int s = 256; s > 0; s >>= 1) {
        if (b < s) red[b] += red[b + s];
        __syncthreads();
    }
    if (b == 0) out[0] = red[0] / (float)BATCH;
}

// ---------------------------------------------------------------------------
// launch: ws layout (all 256B-aligned, ~106.2 MB total):
//   wh      f16 [CPAD x 512]   102,432,768 B
//   xh      f16 [512  x 512]       524,288 B
//   partial f32 [NBLK x 512]     3,201,024 B
//   cost    f32 [512]                2,048 B
// ---------------------------------------------------------------------------
extern "C" void kernel_launch(void* const* d_in, const int* in_sizes, int n_in,
                              void* d_out, int out_size, void* d_ws, size_t ws_size,
                              hipStream_t stream) {
    (void)in_sizes; (void)n_in; (void)out_size; (void)ws_size;
    const float* x   = (const float*)d_in[0];
    const float* w   = (const float*)d_in[1];
    const int*   tgt = (const int*)d_in[2];
    float*       out = (float*)d_out;

    char* ws = (char*)d_ws;
    size_t off = 0;
    _Float16* wh = (_Float16*)(ws + off); off += (size_t)CPAD * EMBED * sizeof(_Float16);
    _Float16* xh = (_Float16*)(ws + off); off += (size_t)BATCH * EMBED * sizeof(_Float16);
    float* partial = (float*)(ws + off);  off += (size_t)NBLK * BATCH * sizeof(float);
    float* cost    = (float*)(ws + off);  off += (size_t)BATCH * sizeof(float);

    arcface_nrm_f16<<<CPAD, 256, 0, stream>>>(w, wh, NCLS);
    arcface_nrm_f16<<<BATCH, 256, 0, stream>>>(x, xh, BATCH);
    arcface_gemm_exp<<<dim3(NBLK, BATCH / MTILE), 256, 0, stream>>>(xh, wh, partial);
    arcface_cost<<<BATCH, 256, 0, stream>>>(xh, wh, tgt, cost);
    arcface_finalize<<<1, 512, 0, stream>>>(partial, cost, out);
}
